// Attention_67138928771342
// MI455X (gfx1250) — compile-verified, matrix operand order checked by reading, not run on hardware
//
#include <hip/hip_runtime.h>

// ---------------------------------------------------------------------------
// Problem constants (from the reference)
// ---------------------------------------------------------------------------
#define DIM      2048
#define N_HEADS  16
#define HEAD_DIM 128
#define BSZ      2
#define SEQLEN   2048
#define ROWS     (BSZ * SEQLEN)          // 4096 total token rows

typedef __attribute__((ext_vector_type(16))) __bf16 v16bf;
typedef __attribute__((ext_vector_type(8)))  float  v8f;

__device__ inline v8f vzero8() {
    v8f z;
    for (int i = 0; i < 8; ++i) z[i] = 0.0f;
    return z;
}

__device__ inline v8f wmma_bf16(v16bf a, v16bf b, v8f c) {
    // D = A(16x32 bf16) x B(32x16 bf16) + C(16x16 f32)
    return __builtin_amdgcn_wmma_f32_16x16x32_bf16(
        /*neg_a=*/false, a, /*neg_b=*/false, b,
        /*c_mod=*/(short)0, c, /*reuse_a=*/false, /*reuse_b=*/false);
}

// ---------------------------------------------------------------------------
// CDNA5 async global->LDS staging (ASYNCcnt-tracked). Per-lane 32B copy as
// two b128 transfers; the 24-bit instruction offset applies to BOTH the LDS
// destination and the global source address (CDNA5 ISA).
// ---------------------------------------------------------------------------
__device__ inline void async_copy32(unsigned int lds_byte, const void* g) {
    asm volatile("global_load_async_to_lds_b128 %0, %1, off\n\t"
                 "global_load_async_to_lds_b128 %0, %1, off offset:16"
                 :: "v"(lds_byte), "v"(g) : "memory");
}
__device__ inline void wait_async_le(int) {      // tile t done, t+1 in flight
    asm volatile("s_wait_asynccnt 0x20" ::: "memory");
}
__device__ inline void wait_async_all() {
    asm volatile("s_wait_asynccnt 0x0" ::: "memory");
}
// Generic (flat) pointer to an LDS object: addr[31:0] is the LDS byte offset.
__device__ inline unsigned int lds_off(const void* p) {
    return (unsigned int)(unsigned long long)p;
}

// ---------------------------------------------------------------------------
// fp32 -> bf16 conversion (grid-stride)
// ---------------------------------------------------------------------------
__global__ void cvt_f32_bf16(const float* __restrict__ in,
                             __bf16* __restrict__ out, int n) {
    int i = blockIdx.x * blockDim.x + threadIdx.x;
    int stride = gridDim.x * blockDim.x;
    for (; i < n; i += stride) out[i] = (__bf16)in[i];
}

// ---------------------------------------------------------------------------
// C[M,N] = A[M,K] * B[N,K]^T   (all row-major, bf16 inputs, f32 accumulate)
// mode 0: store bf16 row-major [M,N]
// mode 1: store bf16 transposed into Vt[b][h][d][s]  (requires M=ROWS, N=DIM)
// mode 2: store f32 row-major [M,N]
// One wave computes a 32x32 tile (2x2 of 16x16 WMMA); block = 4 waves = 64x64.
// Working set (8MB weight panel + x rows) lives in the 192MB L2.
// ---------------------------------------------------------------------------
__device__ inline void store_c(v8f c, void* Cout, int mbase, int nbase,
                               int N, int mode, int lo, int hi) {
    for (int r = 0; r < 8; ++r) {
        int row = mbase + r + 8 * hi;   // C layout: VGPR r -> M=r (+8 for hi half)
        int col = nbase + lo;           // N = lane&15
        float val = c[r];
        if (mode == 0) {
            ((__bf16*)Cout)[(size_t)row * N + col] = (__bf16)val;
        } else if (mode == 1) {
            int bb = row >> 11, s = row & (SEQLEN - 1);
            int hh = col >> 7,  d = col & (HEAD_DIM - 1);
            ((__bf16*)Cout)[(((size_t)(bb * N_HEADS + hh)) * HEAD_DIM + d) * SEQLEN + s]
                = (__bf16)val;
        } else {
            ((float*)Cout)[(size_t)row * N + col] = val;
        }
    }
}

__global__ void gemm_bf16_wmma(const __bf16* __restrict__ A,
                               const __bf16* __restrict__ Bw,
                               void* __restrict__ Cout,
                               int M, int N, int K, int mode) {
    const int lane = threadIdx.x & 31;
    const int wave = threadIdx.x >> 5;     // 0..3
    const int lo = lane & 15;
    const int hi = lane >> 4;
    const int m0 = blockIdx.y * 64 + (wave >> 1) * 32;
    const int n0 = blockIdx.x * 64 + (wave & 1) * 32;

    v8f c00 = vzero8(), c01 = vzero8(), c10 = vzero8(), c11 = vzero8();

    const __bf16* a0p = A  + (size_t)(m0 + lo)      * K + 16 * hi;
    const __bf16* a1p = A  + (size_t)(m0 + 16 + lo) * K + 16 * hi;
    const __bf16* b0p = Bw + (size_t)(n0 + lo)      * K + 16 * hi;
    const __bf16* b1p = Bw + (size_t)(n0 + 16 + lo) * K + 16 * hi;

    for (int k0 = 0; k0 < K; k0 += 32) {
        v16bf a0 = *(const v16bf*)(a0p + k0);
        v16bf a1 = *(const v16bf*)(a1p + k0);
        v16bf b0 = *(const v16bf*)(b0p + k0);
        v16bf b1 = *(const v16bf*)(b1p + k0);
        if (k0 + 256 < K) {
            __builtin_prefetch((const void*)(a0p + k0 + 256), 0, 1);
            __builtin_prefetch((const void*)(b0p + k0 + 256), 0, 1);
        }
        c00 = wmma_bf16(a0, b0, c00);
        c01 = wmma_bf16(a0, b1, c01);
        c10 = wmma_bf16(a1, b0, c10);
        c11 = wmma_bf16(a1, b1, c11);
    }

    store_c(c00, Cout, m0,      n0,      N, mode, lo, hi);
    store_c(c01, Cout, m0,      n0 + 16, N, mode, lo, hi);
    store_c(c10, Cout, m0 + 16, n0,      N, mode, lo, hi);
    store_c(c11, Cout, m0 + 16, n0 + 16, N, mode, lo, hi);
}

// ---------------------------------------------------------------------------
// RoPE applied in-place to Q and K (bf16, layout [b,s,h,d], interleaved pairs)
// ---------------------------------------------------------------------------
__global__ void rope_kernel(__bf16* __restrict__ Q, __bf16* __restrict__ Kb,
                            const float* __restrict__ fcos,
                            const float* __restrict__ fsin) {
    size_t idx = (size_t)blockIdx.x * blockDim.x + threadIdx.x;
    if (idx >= (size_t)ROWS * (DIM / 2)) return;
    int row = (int)(idx >> 10);            // DIM/2 = 1024 pairs per row
    int pr  = (int)(idx & 1023);
    int h = pr >> 6;                       // 64 pairs per head
    int p = pr & 63;
    int s = row & (SEQLEN - 1);
    float c  = fcos[s * 64 + p];
    float sn = fsin[s * 64 + p];
    size_t off = (size_t)row * DIM + h * HEAD_DIM + 2 * p;

    float qa = (float)Q[off], qb = (float)Q[off + 1];
    Q[off]     = (__bf16)(qa * c - qb * sn);
    Q[off + 1] = (__bf16)(qa * sn + qb * c);

    float ka = (float)Kb[off], kb = (float)Kb[off + 1];
    Kb[off]     = (__bf16)(ka * c - kb * sn);
    Kb[off + 1] = (__bf16)(ka * sn + kb * c);
}

// ---------------------------------------------------------------------------
// Flash attention: one wave32 per (b, h, 16-row q-tile).
// Q,K layout [b,s,h,d] bf16; V transposed Vt[b,h,d,s] bf16; O layout [b,s,h,d].
// K/V tiles are staged global->LDS with ASYNC copies, double-buffered so tile
// t+1 streams in while tile t feeds the WMMAs. 16 fragment slots per tile
// (8 K + 8 V), 32B per lane per slot.
// ---------------------------------------------------------------------------
#define NSLOT 16                       // 8 K frags + 8 V frags per k-tile
#define SLOT_ELEMS (32 * 16)           // 32 lanes x 16 bf16 per slot

__global__ void flash_attn(const __bf16* __restrict__ Q,
                           const __bf16* __restrict__ Kb,
                           const __bf16* __restrict__ Vt,
                           __bf16* __restrict__ O) {
    __shared__ __align__(32) __bf16 lds_kv[2][NSLOT * SLOT_ELEMS];  // 2 x 16KB
    __shared__ __align__(32) __bf16 lds_p[16 * 32];                 // 1KB

    const int lane = threadIdx.x & 31;
    const int lo = lane & 15;
    const int hi = lane >> 4;

    const int bid = blockIdx.x;            // b*N_HEADS*128 + h*128 + qt
    const int qt = bid & 127;
    const int h  = (bid >> 7) & (N_HEADS - 1);
    const int b  = bid >> 11;
    const int q0 = qt * 16;
    const float scale = 0.08838834764831845f;   // 1/sqrt(128)

    // Q fragments: 4 chunks over head_dim (4 x 16x32)
    v16bf qf[4];
    const __bf16* qbase = Q + (size_t)(b * SEQLEN + q0 + lo) * DIM
                            + h * HEAD_DIM + 16 * hi;
    for (int dc = 0; dc < 4; ++dc) qf[dc] = *(const v16bf*)(qbase + dc * 32);

    v8f of[8];
    for (int n = 0; n < 8; ++n) of[n] = vzero8();
    float mrow[8], lrow[8];
    for (int r = 0; r < 8; ++r) { mrow[r] = -1e30f; lrow[r] = 0.0f; }

    const __bf16* kbase = Kb + (size_t)(b * SEQLEN) * DIM + h * HEAD_DIM + 16 * hi;
    const __bf16* vbase = Vt + ((size_t)(b * N_HEADS + h) * HEAD_DIM + lo) * SEQLEN
                             + 16 * hi;

    const unsigned int lds_base[2] = { lds_off(&lds_kv[0][0]),
                                       lds_off(&lds_kv[1][0]) };

    const int ktiles = (q0 + 16 + 31) >> 5;    // causal: k <= q0+15

    // Issue async staging for one k-tile into buffer `buf`.
    auto issue_tile = [&](int k0, int buf) {
        unsigned int base = lds_base[buf];
        for (int dc = 0; dc < 4; ++dc)
            for (int j = 0; j < 2; ++j) {
                const void* g = kbase + (size_t)(k0 + 16 * j + lo) * DIM + dc * 32;
                async_copy32(base + (unsigned)(((dc * 2 + j) * 32 + lane) * 32), g);
            }
        for (int n = 0; n < 8; ++n) {
            const void* g = vbase + (size_t)n * 16 * SEQLEN + k0;
            async_copy32(base + (unsigned)(((8 + n) * 32 + lane) * 32), g);
        }
    };

    issue_tile(0, 0);                          // prologue: stage tile 0

    for (int kt = 0; kt < ktiles; ++kt) {
        const int k0 = kt * 32;
        const __bf16* buf = lds_kv[kt & 1];

        // Stage tile t+1 while we compute tile t.
        if (kt + 1 < ktiles) {
            issue_tile(k0 + 32, (kt + 1) & 1);
            wait_async_le(32);                 // tile t complete (in-order)
        } else {
            wait_async_all();
        }

        // ---- S = Q K^T for 32 key columns (two 16-wide N-subtiles) ----
        v8f s0 = vzero8(), s1 = vzero8();
        for (int dc = 0; dc < 4; ++dc) {
            v16bf kf0 = *(const v16bf*)(buf + ((dc * 2 + 0) * 32 + lane) * 16);
            v16bf kf1 = *(const v16bf*)(buf + ((dc * 2 + 1) * 32 + lane) * 16);
            s0 = wmma_bf16(qf[dc], kf0, s0);
            s1 = wmma_bf16(qf[dc], kf1, s1);
        }

        // ---- causal mask + scale; per-row online softmax stats ----
        float p0[8], p1[8], tmax[8];
        for (int r = 0; r < 8; ++r) {
            int q  = q0 + r + 8 * hi;
            int kA = k0 + lo;
            int kB = k0 + 16 + lo;
            float v0 = (kA <= q) ? s0[r] * scale : -1e30f;
            float v1 = (kB <= q) ? s1[r] * scale : -1e30f;
            p0[r] = v0; p1[r] = v1;
            tmax[r] = fmaxf(v0, v1);
        }
        for (int d = 1; d < 16; d <<= 1)
            for (int r = 0; r < 8; ++r)
                tmax[r] = fmaxf(tmax[r], __shfl_xor(tmax[r], d, 16));

        float sc[8];
        for (int r = 0; r < 8; ++r) {
            float mnew = fmaxf(mrow[r], tmax[r]);
            sc[r] = __expf(mrow[r] - mnew);
            mrow[r] = mnew;
            p0[r] = __expf(p0[r] - mnew);
            p1[r] = __expf(p1[r] - mnew);
        }
        float tsum[8];
        for (int r = 0; r < 8; ++r) tsum[r] = p0[r] + p1[r];
        for (int d = 1; d < 16; d <<= 1)
            for (int r = 0; r < 8; ++r)
                tsum[r] += __shfl_xor(tsum[r], d, 16);
        for (int r = 0; r < 8; ++r) lrow[r] = lrow[r] * sc[r] + tsum[r];

        for (int n = 0; n < 8; ++n)
            for (int r = 0; r < 8; ++r)
                of[n][r] *= sc[r];

        // ---- transpose P (C-layout -> A-layout) through LDS ----
        __syncthreads();   // WAR vs previous iteration's fragment read
        for (int r = 0; r < 8; ++r) {
            int row = r + 8 * hi;
            lds_p[row * 32 + lo]      = (__bf16)p0[r];
            lds_p[row * 32 + 16 + lo] = (__bf16)p1[r];
        }
        __syncthreads();   // RAW before A-fragment load
        v16bf pf = *(const v16bf*)(lds_p + lo * 32 + 16 * hi);

        // ---- O += P V : 8 N-chunks of 16 over head_dim ----
        for (int n = 0; n < 8; ++n) {
            v16bf vf = *(const v16bf*)(buf + ((8 + n) * 32 + lane) * 16);
            of[n] = wmma_bf16(pf, vf, of[n]);
        }
    }

    // ---- epilogue: divide by l, store bf16 O[b,s,h,d] ----
    __bf16* obase = O + (size_t)(b * SEQLEN + q0) * DIM + h * HEAD_DIM;
    for (int n = 0; n < 8; ++n) {
        for (int r = 0; r < 8; ++r) {
            int row = r + 8 * hi;
            float val = of[n][r] / lrow[r];
            obase[(size_t)row * DIM + n * 16 + lo] = (__bf16)val;
        }
    }
}

// ---------------------------------------------------------------------------
// Launch
// ---------------------------------------------------------------------------
extern "C" void kernel_launch(void* const* d_in, const int* in_sizes, int n_in,
                              void* d_out, int out_size, void* d_ws, size_t ws_size,
                              hipStream_t stream) {
    (void)in_sizes; (void)n_in; (void)out_size; (void)ws_size;

    const float* x    = (const float*)d_in[0];
    const float* fcos = (const float*)d_in[2];
    const float* fsin = (const float*)d_in[3];
    const float* wq   = (const float*)d_in[5];
    const float* wk   = (const float*)d_in[6];
    const float* wv   = (const float*)d_in[7];
    const float* wo   = (const float*)d_in[8];
    float* out = (float*)d_out;

    // Workspace layout (bf16 elements). Total = 112 MB < 192 MB L2.
    const size_t RD = (size_t)ROWS * DIM;   // 8M elems
    const size_t WW = (size_t)DIM * DIM;    // 4M elems
    __bf16* xb  = (__bf16*)d_ws;
    __bf16* wqb = xb  + RD;
    __bf16* wkb = wqb + WW;
    __bf16* wvb = wkb + WW;
    __bf16* wob = wvb + WW;
    __bf16* Qb  = wob + WW;
    __bf16* Kbf = Qb  + RD;
    __bf16* Vt  = Kbf + RD;
    __bf16* Ob  = Vt  + RD;

    // 1) fp32 -> bf16 conversions
    cvt_f32_bf16<<<8192, 256, 0, stream>>>(x,  xb,  (int)RD);
    cvt_f32_bf16<<<4096, 256, 0, stream>>>(wq, wqb, (int)WW);
    cvt_f32_bf16<<<4096, 256, 0, stream>>>(wk, wkb, (int)WW);
    cvt_f32_bf16<<<4096, 256, 0, stream>>>(wv, wvb, (int)WW);
    cvt_f32_bf16<<<4096, 256, 0, stream>>>(wo, wob, (int)WW);

    // 2) QKV projections (WMMA GEMMs). V is stored transposed per-head.
    dim3 gblk(128);
    dim3 ggrid(DIM / 64, ROWS / 64);  // (32, 64)
    gemm_bf16_wmma<<<ggrid, gblk, 0, stream>>>(xb, wqb, Qb,  ROWS, DIM, DIM, 0);
    gemm_bf16_wmma<<<ggrid, gblk, 0, stream>>>(xb, wkb, Kbf, ROWS, DIM, DIM, 0);
    gemm_bf16_wmma<<<ggrid, gblk, 0, stream>>>(xb, wvb, Vt,  ROWS, DIM, DIM, 1);

    // 3) RoPE on Q and K
    rope_kernel<<<(ROWS * (DIM / 2)) / 256, 256, 0, stream>>>(Qb, Kbf, fcos, fsin);

    // 4) Flash attention: one wave per (b, h, q-tile of 16), async K/V staging
    flash_attn<<<BSZ * N_HEADS * (SEQLEN / 16), 32, 0, stream>>>(Qb, Kbf, Vt, Ob);

    // 5) Output projection, fp32 result
    gemm_bf16_wmma<<<ggrid, gblk, 0, stream>>>(Ob, wob, out, ROWS, DIM, DIM, 2);
}
